// MultilayerGRU_48773648613827
// MI455X (gfx1250) — matrix operand order repaired and checked by reading, not compiled, not run on hardware
//
#include <hip/hip_runtime.h>

// MI455X (gfx1250) multilayer GRU.
// Strategy:
//   * Convert all weights fp32->bf16 once (26 MB, L2-resident thereafter).
//   * Precompute layer-0 input projections x @ Wx{z,r,g}[0]^T for all B*S rows
//     as one large WMMA GEMM (off the serial critical path).
//   * Persistent kernel (64 blocks x 256 thr = 512 wave32 waves) walks the 512
//     timesteps with a device-wide sense-reversing barrier; 4 barriers/step.
//   * All matmuls use v_wmma_f32_16x16x32_bf16 (f32 accum). A/B tiles are
//     loaded as two contiguous b128 per lane matching the ISA VGPR layouts.
//   * Each 16x16 output tile has exactly one producer wave (deterministic).
// Workspace requirement: ~129 MB.

typedef __bf16 bf16_t;
typedef __attribute__((ext_vector_type(16))) __bf16 v16bf;
typedef __attribute__((ext_vector_type(8)))  float  v8f;

constexpr int    kB   = 32;
constexpr int    kS   = 512;
constexpr int    kH   = 1024;
constexpr int    kL   = 2;
constexpr int    kBH  = kB * kH;
constexpr size_t kBSH = (size_t)kB * kS * kH;
constexpr int    kKS  = kH / 32;   // 32 K-steps of depth 32
constexpr int    kNT  = kH / 16;   // 64 column tiles
constexpr int    NBLK = 64;
constexpr int    NTHR = 256;
constexpr int    NWAVE = NBLK * NTHR / 32;  // 512 waves

struct GruParams {
  const float* x;
  const float* h0;
  const float* bhz; const float* bhr; const float* bhg; const float* bY;
  const bf16_t* Wxz; const bf16_t* Wxr; const bf16_t* Wxg;
  const bf16_t* Whz; const bf16_t* Whr; const bf16_t* Whg;
  const bf16_t* WY;
  const bf16_t* xproj;   // [3][B*S][H] bf16, gate order z,r,g
  float*  hf32;          // [L][B*H]
  bf16_t* hbf;           // [L][B*H]
  float*  zbuf;          // [L][B*H]
  bf16_t* rhbuf;         // [L][B*H]
  float*  gxbuf;         // [L][B*H]
  float*  out;           // [B*S*H] layer_output then [B*L*H] hidden_state
  unsigned* bar;         // [0]=count [1]=generation
};

// ---------------- WMMA tile loaders (per CDNA5 ISA 7.12.2 layouts) ----------

// A tile (16x32 bf16, M rows x K): lane = half*16+m holds row m.
// elems 0..7  -> K = kb + half*8 + 0..7
// elems 8..15 -> K = kb + 16 + half*8 + 0..7
__device__ inline v16bf load_a_bf16(const bf16_t* X, int ldx, int mb, int kb, int lane) {
  int m = lane & 15, half = lane >> 4;
  const bf16_t* p = X + (size_t)(mb + m) * ldx + kb + half * 8;
  v16bf a;
  *reinterpret_cast<float4*>(&a)       = *reinterpret_cast<const float4*>(p);
  *(reinterpret_cast<float4*>(&a) + 1) = *reinterpret_cast<const float4*>(p + 16);
  return a;
}

// Same A layout but reading fp32 source with convert.
__device__ inline v16bf load_a_f32cvt(const float* X, int ldx, int mb, int kb, int lane) {
  int m = lane & 15, half = lane >> 4;
  const float* p = X + (size_t)(mb + m) * ldx + kb + half * 8;
  v16bf a;
#pragma unroll
  for (int i = 0; i < 8; ++i) a[i] = (bf16_t)p[i];
#pragma unroll
  for (int i = 0; i < 8; ++i) a[8 + i] = (bf16_t)p[16 + i];
  return a;
}

// B tile (32x16 bf16, K x N) taken from row-major W so B[K][N] = W[nb+N][kb+K]:
// lane = half*16+n holds column n; elems 0..7 -> K = kb + half*8 + 0..7, elems 8..15 -> K = kb + 16 + half*8 + 0..7.
__device__ inline v16bf load_b_bf16(const bf16_t* W, int ldw, int nb, int kb, int lane) {
  int n = lane & 15, half = lane >> 4;
  const bf16_t* p = W + (size_t)(nb + n) * ldw + kb + half * 8;
  v16bf b;
  *reinterpret_cast<float4*>(&b)       = *reinterpret_cast<const float4*>(p);
  *(reinterpret_cast<float4*>(&b) + 1) = *reinterpret_cast<const float4*>(p + 16);
  return b;
}

__device__ inline v8f wmma_bf16(v8f c, v16bf a, v16bf b) {
  return __builtin_amdgcn_wmma_f32_16x16x32_bf16(false, a, false, b, (short)0, c,
                                                 false, false);
}

// Full K=1024 accumulation of A[32x1024] row-block mb against W (y = A*W^T tile).
__device__ inline v8f gemm_acc(v8f c, const bf16_t* A, const bf16_t* W,
                               int mb, int nb, int lane) {
#pragma unroll 4
  for (int ks = 0; ks < kKS; ++ks) {
    v16bf a = load_a_bf16(A, kH, mb, ks * 32, lane);
    v16bf b = load_b_bf16(W, kH, nb, ks * 32, lane);
    c = wmma_bf16(c, a, b);
  }
  return c;
}

__device__ inline float fsigmoid(float x) { return 1.f / (1.f + __expf(-x)); }
__device__ inline float ftanh(float x) {
  float e = __expf(-2.f * x);
  return (1.f - e) / (1.f + e);
}

// ---------------- device-wide sense-reversing barrier -----------------------

__device__ inline void grid_barrier(unsigned* bar) {
  __threadfence();   // publish this block's stores at agent scope
  __syncthreads();
  if (threadIdx.x == 0) {
    unsigned* cnt = bar;
    unsigned* gen = bar + 1;
    unsigned g = __hip_atomic_load(gen, __ATOMIC_RELAXED, __HIP_MEMORY_SCOPE_AGENT);
    unsigned a = __hip_atomic_fetch_add(cnt, 1u, __ATOMIC_ACQ_REL,
                                        __HIP_MEMORY_SCOPE_AGENT);
    if (a == (unsigned)NBLK - 1u) {
      __hip_atomic_store(cnt, 0u, __ATOMIC_RELAXED, __HIP_MEMORY_SCOPE_AGENT);
      __hip_atomic_store(gen, g + 1u, __ATOMIC_RELEASE, __HIP_MEMORY_SCOPE_AGENT);
    } else {
      while (__hip_atomic_load(gen, __ATOMIC_ACQUIRE, __HIP_MEMORY_SCOPE_AGENT) == g)
        __builtin_amdgcn_s_sleep(2);
    }
  }
  __syncthreads();
}

// ---------------- simple fp32 -> bf16 conversion ----------------------------

__global__ __launch_bounds__(256) void cvt_f32_bf16(const float* in, bf16_t* out,
                                                    size_t n) {
  size_t i = blockIdx.x * (size_t)blockDim.x + threadIdx.x;
  size_t st = (size_t)gridDim.x * blockDim.x;
  for (; i < n; i += st) out[i] = (bf16_t)in[i];
}

// ---------------- big precompute GEMM: xproj[g] = x @ Wx{g}[0]^T ------------

__global__ __launch_bounds__(NTHR) void xproj_kernel(const float* __restrict__ x,
                                                     const bf16_t* __restrict__ Wz,
                                                     const bf16_t* __restrict__ Wr,
                                                     const bf16_t* __restrict__ Wg,
                                                     bf16_t* __restrict__ xproj) {
  const int lane  = threadIdx.x & 31;
  const int wave  = (int)((blockIdx.x * blockDim.x + threadIdx.x) >> 5);
  const int waves = (int)((gridDim.x * blockDim.x) >> 5);
  const int mtiles = (kB * kS) / 16;         // 1024
  const int total  = 3 * mtiles * kNT;       // 196608 tile jobs
  for (int job = wave; job < total; job += waves) {
    int gate = job / (mtiles * kNT);
    int rem  = job % (mtiles * kNT);
    int mt = rem / kNT, nt = rem % kNT;
    int mb = mt * 16, nb = nt * 16;
    const bf16_t* W = (gate == 0) ? Wz : (gate == 1) ? Wr : Wg;
    v8f c = {0.f, 0.f, 0.f, 0.f, 0.f, 0.f, 0.f, 0.f};
#pragma unroll 4
    for (int ks = 0; ks < kKS; ++ks) {
      v16bf a = load_a_f32cvt(x, kH, mb, ks * 32, lane);
      v16bf b = load_b_bf16(W, kH, nb, ks * 32, lane);
      c = wmma_bf16(c, a, b);
    }
    int n = nb + (lane & 15), half = lane >> 4;
    bf16_t* op = xproj + (size_t)gate * kBSH;
#pragma unroll
    for (int v = 0; v < 8; ++v) {
      int m = mb + half * 8 + v;
      op[(size_t)m * kH + n] = (bf16_t)c[v];
    }
  }
}

// ---------------- persistent recurrent kernel -------------------------------

// Phase A job (per layer k): cls 0 = z, 1 = r (+r*h), 2 = x-side of g.
__device__ void phaseA_job(const GruParams& p, int k, int t, int job, int lane) {
  int cls  = job >> 7;            // 0,1,2
  int tile = job & 127;
  int mb = (tile >> 6) * 16, nb = (tile & 63) * 16;
  int n = nb + (lane & 15), half = lane >> 4;

  if (cls == 2 && k == 0) {
    // layer0 g x-side is fully precomputed: gx = xproj_g + bhg[0]
    const bf16_t* xg = p.xproj + 2 * kBSH;
#pragma unroll
    for (int v = 0; v < 8; ++v) {
      int m = mb + half * 8 + v;
      p.gxbuf[(size_t)k * kBH + (size_t)m * kH + n] =
          (float)xg[((size_t)m * kS + t) * kH + n] + p.bhg[n];
    }
    return;
  }

  const bf16_t* Wh = nullptr;
  const bf16_t* Wx = nullptr;
  const float*  bias;
  size_t WHH = (size_t)kH * kH;
  if (cls == 0)      { Wh = p.Whz + k * WHH; Wx = p.Wxz + WHH; bias = p.bhz + k * kH; }
  else if (cls == 1) { Wh = p.Whr + k * WHH; Wx = p.Wxr + WHH; bias = p.bhr + k * kH; }
  else               { /* k==1 g x-side */    Wx = p.Wxg + WHH; bias = p.bhg + kH; }

  v8f c = {0.f, 0.f, 0.f, 0.f, 0.f, 0.f, 0.f, 0.f};
  if (Wh) c = gemm_acc(c, p.hbf + (size_t)k * kBH, Wh, mb, nb, lane);
  if (k == 1) c = gemm_acc(c, p.hbf /* layer0 h = x1 */, Wx, mb, nb, lane);

#pragma unroll
  for (int v = 0; v < 8; ++v) {
    int m = mb + half * 8 + v;
    size_t idx = (size_t)k * kBH + (size_t)m * kH + n;
    float val = c[v] + bias[n];
    if (k == 0)
      val += (float)p.xproj[(size_t)cls * kBSH + ((size_t)m * kS + t) * kH + n];
    if (cls == 0) {
      p.zbuf[idx] = fsigmoid(val);
    } else if (cls == 1) {
      float r = fsigmoid(val);
      p.rhbuf[idx] = (bf16_t)(r * p.hf32[idx]);
    } else {
      p.gxbuf[idx] = val;
    }
  }
}

// Phase B job (per layer k): g = tanh((r*h)@Whg^T + gx); h = z*h + (1-z)*g.
__device__ void phaseB_job(const GruParams& p, int k, int tile, int lane) {
  int mb = (tile >> 6) * 16, nb = (tile & 63) * 16;
  int n = nb + (lane & 15), half = lane >> 4;
  v8f c = {0.f, 0.f, 0.f, 0.f, 0.f, 0.f, 0.f, 0.f};
  c = gemm_acc(c, p.rhbuf + (size_t)k * kBH, p.Whg + (size_t)k * kH * kH, mb, nb, lane);
#pragma unroll
  for (int v = 0; v < 8; ++v) {
    int m = mb + half * 8 + v;
    size_t idx = (size_t)k * kBH + (size_t)m * kH + n;
    float g  = ftanh(c[v] + p.gxbuf[idx]);
    float z  = p.zbuf[idx];
    float hn = z * p.hf32[idx] + (1.f - z) * g;
    p.hf32[idx] = hn;
    p.hbf[idx]  = (bf16_t)hn;
  }
}

// Output projection for timestep t: out[b][t][:] = h1 @ WY^T + bY.
__device__ void y_job(const GruParams& p, int t, int tile, int lane) {
  int mb = (tile >> 6) * 16, nb = (tile & 63) * 16;
  int n = nb + (lane & 15), half = lane >> 4;
  v8f c = {0.f, 0.f, 0.f, 0.f, 0.f, 0.f, 0.f, 0.f};
  c = gemm_acc(c, p.hbf + kBH /* layer1 h */, p.WY, mb, nb, lane);
#pragma unroll
  for (int v = 0; v < 8; ++v) {
    int m = mb + half * 8 + v;   // batch row
    p.out[((size_t)m * kS + t) * kH + n] = c[v] + p.bY[n];
  }
}

__global__ __launch_bounds__(NTHR) void gru_persistent(GruParams p) {
  const int lane = threadIdx.x & 31;
  const int wave = (int)((blockIdx.x * blockDim.x + threadIdx.x) >> 5);

  // init h state from h0 ([L,B,H] flat matches [L][B*H])
  for (int i = (int)(blockIdx.x * blockDim.x + threadIdx.x); i < kL * kBH;
       i += NBLK * NTHR) {
    float v = p.h0[i];
    p.hf32[i] = v;
    p.hbf[i]  = (bf16_t)v;
  }
  grid_barrier(p.bar);

  for (int t = 0; t < kS; ++t) {
    // P1: layer0 {z, r, gx-copy} plus previous step's output GEMM
    int nj = (t > 0) ? 512 : 384;
    for (int job = wave; job < nj; job += NWAVE) {
      if (job >= 384) y_job(p, t - 1, job - 384, lane);
      else            phaseA_job(p, 0, t, job, lane);
    }
    grid_barrier(p.bar);
    // P2: layer0 g + h-update
    for (int job = wave; job < 128; job += NWAVE) phaseB_job(p, 0, job, lane);
    grid_barrier(p.bar);
    // P3: layer1 {z, r, gx-GEMM}
    for (int job = wave; job < 384; job += NWAVE) phaseA_job(p, 1, t, job, lane);
    grid_barrier(p.bar);
    // P4: layer1 g + h-update
    for (int job = wave; job < 128; job += NWAVE) phaseB_job(p, 1, job, lane);
    grid_barrier(p.bar);
  }

  // final output GEMM for t = S-1, plus hidden_state [B][L][H]
  for (int job = wave; job < 128; job += NWAVE) y_job(p, kS - 1, job, lane);
  for (int i = (int)(blockIdx.x * blockDim.x + threadIdx.x); i < kB * kL * kH;
       i += NBLK * NTHR) {
    int b = i / (kL * kH), k = (i / kH) % kL, hh = i % kH;
    p.out[(size_t)kB * kS * kH + i] = p.hf32[(size_t)k * kBH + (size_t)b * kH + hh];
  }
}

// ---------------- launch ----------------------------------------------------

extern "C" void kernel_launch(void* const* d_in, const int* in_sizes, int n_in,
                              void* d_out, int out_size, void* d_ws, size_t ws_size,
                              hipStream_t stream) {
  const float* x   = (const float*)d_in[0];
  const float* h0  = (const float*)d_in[1];
  const float* Wxz = (const float*)d_in[2];
  const float* Wxr = (const float*)d_in[3];
  const float* Wxg = (const float*)d_in[4];
  const float* Whz = (const float*)d_in[5];
  const float* Whr = (const float*)d_in[6];
  const float* Whg = (const float*)d_in[7];
  const float* bhz = (const float*)d_in[8];
  const float* bhr = (const float*)d_in[9];
  const float* bhg = (const float*)d_in[10];
  const float* WY  = (const float*)d_in[11];
  const float* bY  = (const float*)d_in[12];

  // workspace carve (needs ~129 MB)
  char* base = (char*)d_ws;
  size_t off = 0;
  auto take = [&](size_t bytes) -> char* {
    char* r = base + off;
    off = (off + bytes + 255) & ~(size_t)255;
    return r;
  };
  const size_t WHH = (size_t)kH * kH;
  bf16_t* wWxz = (bf16_t*)take(kL * WHH * sizeof(bf16_t));
  bf16_t* wWxr = (bf16_t*)take(kL * WHH * sizeof(bf16_t));
  bf16_t* wWxg = (bf16_t*)take(kL * WHH * sizeof(bf16_t));
  bf16_t* wWhz = (bf16_t*)take(kL * WHH * sizeof(bf16_t));
  bf16_t* wWhr = (bf16_t*)take(kL * WHH * sizeof(bf16_t));
  bf16_t* wWhg = (bf16_t*)take(kL * WHH * sizeof(bf16_t));
  bf16_t* wWY  = (bf16_t*)take(WHH * sizeof(bf16_t));
  bf16_t* xproj = (bf16_t*)take(3 * kBSH * sizeof(bf16_t));
  float*  hf32  = (float*)take(kL * kBH * sizeof(float));
  bf16_t* hbf   = (bf16_t*)take(kL * kBH * sizeof(bf16_t));
  float*  zbuf  = (float*)take(kL * kBH * sizeof(float));
  bf16_t* rhbuf = (bf16_t*)take(kL * kBH * sizeof(bf16_t));
  float*  gxbuf = (float*)take(kL * kBH * sizeof(float));
  unsigned* bar = (unsigned*)take(256);

  hipMemsetAsync(bar, 0, 256, stream);

  // weight fp32 -> bf16 conversions
  cvt_f32_bf16<<<1024, 256, 0, stream>>>(Wxz, wWxz, kL * WHH);
  cvt_f32_bf16<<<1024, 256, 0, stream>>>(Wxr, wWxr, kL * WHH);
  cvt_f32_bf16<<<1024, 256, 0, stream>>>(Wxg, wWxg, kL * WHH);
  cvt_f32_bf16<<<1024, 256, 0, stream>>>(Whz, wWhz, kL * WHH);
  cvt_f32_bf16<<<1024, 256, 0, stream>>>(Whr, wWhr, kL * WHH);
  cvt_f32_bf16<<<1024, 256, 0, stream>>>(Whg, wWhg, kL * WHH);
  cvt_f32_bf16<<<1024, 256, 0, stream>>>(WY,  wWY,  WHH);

  // hoisted layer-0 input projections (full-throughput GEMM)
  xproj_kernel<<<2048, NTHR, 0, stream>>>(x, wWxz, wWxr, wWxg, xproj);

  GruParams P;
  P.x = x; P.h0 = h0;
  P.bhz = bhz; P.bhr = bhr; P.bhg = bhg; P.bY = bY;
  P.Wxz = wWxz; P.Wxr = wWxr; P.Wxg = wWxg;
  P.Whz = wWhz; P.Whr = wWhr; P.Whg = wWhg; P.WY = wWY;
  P.xproj = xproj;
  P.hf32 = hf32; P.hbf = hbf; P.zbuf = zbuf; P.rhbuf = rhbuf; P.gxbuf = gxbuf;
  P.out = (float*)d_out;
  P.bar = bar;

  gru_persistent<<<NBLK, NTHR, 0, stream>>>(P);
}
